// ContrastiveLoss_52939766890942
// MI455X (gfx1250) — matrix operand order repaired and checked by reading, not compiled, not run on hardware
//
#include <hip/hip_runtime.h>
#include <hip/hip_bf16.h>

// NT-Xent contrastive loss for MI455X (gfx1250), wave32 + WMMA f16->f32.
//
// d_in[0]: z1 [4096,128] f32   d_in[1]: z2 [4096,128] f32
// d_out  : scalar f32 loss
// d_ws   : [0, 2.03MB)  zn*sqrt(10) as _Float16 row-major [8320][128]
//                       (rows 8192..8319 are prefetch padding, never consumed)
//          then 512 per-block partial sums (f32)

typedef __attribute__((ext_vector_type(16))) _Float16 v16h;
typedef __attribute__((ext_vector_type(8)))  _Float16 v8h;
typedef __attribute__((ext_vector_type(4)))  _Float16 v4h;
typedef __attribute__((ext_vector_type(8)))  float    v8f;

#define NROW    4096
#define MROW    8192
#define DDIM    128
#define NTILE   512                  // MROW / 16
#define PADROWS 128                  // one prefetch stride (8 tiles * 16 rows)
#define SQRT_INV_T 3.16227766016838f // sqrt(1/TEMPERATURE) = sqrt(10)

// ---------------------------------------------------------------------------
// Kernel 1: normalize rows of z = concat(z1, z2), scale by sqrt(10), store f16.
// One wave per row (8 rows / 256-thread block). The sqrt(10) scale makes the
// WMMA output directly equal cos_sim / T, removing per-element muls later.
// ---------------------------------------------------------------------------
__global__ __launch_bounds__(256)
void ntx_normalize(const float* __restrict__ z1, const float* __restrict__ z2,
                   _Float16* __restrict__ zh) {
  const int wave = threadIdx.x >> 5;
  const int lane = threadIdx.x & 31;
  const int row  = blockIdx.x * 8 + wave;

  const float* src = (row < NROW) ? (z1 + (size_t)row * DDIM)
                                  : (z2 + (size_t)(row - NROW) * DDIM);
  float4 x = ((const float4*)src)[lane];            // 4 elems per lane (128/32)
  float ss = x.x * x.x + x.y * x.y + x.z * x.z + x.w * x.w;
#pragma unroll
  for (int off = 16; off > 0; off >>= 1) ss += __shfl_xor(ss, off);

  float nrm = fmaxf(sqrtf(ss), 1e-8f);              // torch CosineSimilarity eps
  const float inv = SQRT_INV_T / nrm;

  v4h o;
  o[0] = (_Float16)(x.x * inv);
  o[1] = (_Float16)(x.y * inv);
  o[2] = (_Float16)(x.z * inv);
  o[3] = (_Float16)(x.w * inv);
  *(v4h*)(zh + (size_t)row * DDIM + lane * 4) = o;
}

// ---------------------------------------------------------------------------
// Kernel 2: per 16-row tile, compute sim row block via WMMA, accumulate
// sum_{c != r} exp(sim[r,c]) and capture pos = sim[r, partner(r)].
// Block = 256 threads = 8 waves; wave w handles column tiles w, w+8, ...
// Double-buffered B fragments overlap global loads with the WMMA pipe.
// Fully deterministic: no atomics, fixed-order LDS reductions.
// ---------------------------------------------------------------------------
__global__ __launch_bounds__(256)
void ntx_sim_rowsum(const _Float16* __restrict__ zh, float* __restrict__ partial) {
  __shared__ float part[8][8][32];   // [wave][acc slot v][lane]
  __shared__ float posLDS[16];       // sim[r, partner(r)]/T per row in tile
  __shared__ float lossArr[16];

  const int wave = __builtin_amdgcn_readfirstlane(threadIdx.x >> 5); // scalar
  const int lane = threadIdx.x & 31;
  const int rt   = blockIdx.x;       // row tile index (0..511)
  const int r0   = rt * 16;
  const int lrow = lane & 15;        // row-in-tile for A, col-in-tile for B/C
  const int hi   = lane >> 4;

  // --- A fragments: 16x32 f16, contiguous k order, same as B.
  // Lane group lo/hi: halves 0..15 = K(koff..koff+15),
  // koff = 16*hi. Two contiguous 16-byte chunks per lane.
  v16h A[4];
  {
    const _Float16* base = zh + (size_t)(r0 + lrow) * DDIM + hi * 16;
#pragma unroll
    for (int kt = 0; kt < 4; ++kt) {
      const _Float16* p = base + kt * 32;
      v8h lo = *(const v8h*)(p);
      v8h hh = *(const v8h*)(p + 8);
#pragma unroll
      for (int e = 0; e < 8; ++e) { A[kt][e] = lo[e]; A[kt][e + 8] = hh[e]; }
    }
  }

  bool mine[8];                      // lane owns C element (Mr, N=lrow) diag?
#pragma unroll
  for (int v = 0; v < 8; ++v) mine[v] = (lrow == v + 8 * hi);

  float acc[8], pos[8];
#pragma unroll
  for (int v = 0; v < 8; ++v) { acc[v] = 0.0f; pos[v] = 0.0f; }

  const int ptile = (rt + (NTILE / 2)) & (NTILE - 1);  // partner column tile
  const int koffB = hi * 16;
  const size_t bstep = (size_t)128 * DDIM;             // +8 tiles (=128 rows)

  // --- B fragment: 32x16 f16. Lane holds column c0+lrow; halves e=0..15 map
  // to K = koffB + e (one contiguous 32-byte chunk, 32B-aligned).
  const _Float16* bptr = zh + (size_t)(wave * 16 + lrow) * DDIM + koffB;

  v16h Bc[4];
#pragma unroll
  for (int kt = 0; kt < 4; ++kt) Bc[kt] = *(const v16h*)(bptr + kt * 32);

  for (int ct = wave; ct < NTILE; ct += 8) {
    // Prefetch next tile's B into fresh registers (pad rows make this safe
    // on the final iteration; those values are never consumed).
    const _Float16* nptr = bptr + bstep;
    v16h Bn[4];
#pragma unroll
    for (int kt = 0; kt < 4; ++kt) Bn[kt] = *(const v16h*)(nptr + kt * 32);

    v8f c = {0.f, 0.f, 0.f, 0.f, 0.f, 0.f, 0.f, 0.f};
#pragma unroll
    for (int kt = 0; kt < 4; ++kt) {
      c = __builtin_amdgcn_wmma_f32_16x16x32_f16(
              /*neg_a=*/false, A[kt], /*neg_b=*/false, Bc[kt],
              /*c_mod=*/(short)0, c, /*reuse_a=*/false, /*reuse_b=*/false);
    }

    const bool isD = (ct == rt);
    const bool isP = (ct == ptile);
#pragma unroll
    for (int v = 0; v < 8; ++v) {
      const float s = c[v];                          // already cos/T (sqrt10^2)
      pos[v] = (isP && mine[v]) ? s : pos[v];        // v_cndmask, branch-free
      const float e = __expf(s);
      acc[v] += (isD && mine[v]) ? 0.0f : e;         // exclude diagonal only
    }

#pragma unroll
    for (int kt = 0; kt < 4; ++kt) Bc[kt] = Bn[kt];
    bptr = nptr;
  }

  // One wave owns the partner tile; its diagonal lanes publish pos (16 unique
  // single writers -> deterministic).
  if ((ptile & 7) == wave) {
#pragma unroll
    for (int v = 0; v < 8; ++v)
      if (mine[v]) posLDS[v + 8 * hi] = pos[v];
  }

#pragma unroll
  for (int v = 0; v < 8; ++v) part[wave][v][lane] = acc[v];
  __syncthreads();

  const int t = threadIdx.x;
  if (t < 16) {
    const int v = t & 7;
    const int h = t >> 3;                       // row t = v + 8*h
    float S = 0.0f;                             // fixed-order reduction
    for (int w = 0; w < 8; ++w)
      for (int j = 0; j < 16; ++j)
        S += part[w][v][h * 16 + j];
    const float p2 = 2.0f * posLDS[t];          // pos = sim[r,p] + sim[p,r]
    lossArr[t] = __logf(__expf(p2) + S) - p2;   // -log_softmax(...)[0]
  }
  __syncthreads();
  if (t == 0) {
    float bs = 0.0f;
    for (int i = 0; i < 16; ++i) bs += lossArr[i];
    partial[rt] = bs;
  }
}

// ---------------------------------------------------------------------------
// Kernel 3: deterministic tree reduction of 512 partials -> scalar loss.
// ---------------------------------------------------------------------------
__global__ __launch_bounds__(512)
void ntx_reduce(const float* __restrict__ partial, float* __restrict__ out) {
  __shared__ float red[512];
  const int t = threadIdx.x;
  red[t] = partial[t];
  __syncthreads();
  for (int s = 256; s > 0; s >>= 1) {
    if (t < s) red[t] += red[t + s];
    __syncthreads();
  }
  if (t == 0) out[0] = red[0] * (1.0f / (8192.0f * 8192.0f));  // mean/M = sum/M^2
}

extern "C" void kernel_launch(void* const* d_in, const int* in_sizes, int n_in,
                              void* d_out, int out_size, void* d_ws, size_t ws_size,
                              hipStream_t stream) {
  (void)in_sizes; (void)n_in; (void)out_size; (void)ws_size;
  const float* z1 = (const float*)d_in[0];
  const float* z2 = (const float*)d_in[1];
  float* out = (float*)d_out;

  _Float16* zh = (_Float16*)d_ws;
  float* partial =
      (float*)((char*)d_ws +
               (size_t)(MROW + PADROWS) * DDIM * sizeof(_Float16));

  ntx_normalize<<<MROW / 8, 256, 0, stream>>>(z1, z2, zh);
  ntx_sim_rowsum<<<NTILE, 256, 0, stream>>>(zh, partial);
  ntx_reduce<<<1, 512, 0, stream>>>(partial, out);
}
